// MixedSparseFullAttention_11321533792752
// MI455X (gfx1250) — matrix operand order repaired and checked by reading, not compiled, not run on hardware
//
#include <hip/hip_runtime.h>
#include <hip/hip_bf16.h>

typedef __attribute__((ext_vector_type(16))) _Float16 v16h;
typedef __attribute__((ext_vector_type(8)))  float    v8f;
typedef __attribute__((ext_vector_type(4)))  int      v4i;

#define H   2048
#define SQ  2048
#define BB  2
#define MROWS (BB*SQ)      // 4096
#define HEADS 16
#define HD  128
#define RLORA 16

// ---- CDNA5 async global->LDS path (guarded; falls back to load+ds_store) ----
#if defined(__has_builtin)
#if __has_builtin(__builtin_amdgcn_global_load_async_to_lds_b128) && \
    __has_builtin(__builtin_amdgcn_s_wait_asynccnt)
#define USE_ASYNC_LDS 1
#endif
#endif

#ifdef USE_ASYNC_LDS
static __device__ inline void async_ld_b128(const void* g, void* l) {
    // Builtin signature (from hipcc diagnostic): arg0 = addrspace(1) v4i*,
    // arg1 = LDS v4i*. LDS aperture keeps the byte offset in addr[31:0]
    // (ISA 10.2), so a 32-bit truncation of the flat address is the
    // addrspace(3) pointer.
    __builtin_amdgcn_global_load_async_to_lds_b128(
        (__attribute__((address_space(1))) v4i*)(uintptr_t)g,
        (__attribute__((address_space(3))) v4i*)(unsigned)(uintptr_t)l,
        0, 0);
}
static __device__ inline void async_wait() {
    __builtin_amdgcn_s_wait_asynccnt(0);
}
#else
static __device__ inline void async_ld_b128(const void* g, void* l) {
    *(uint4*)l = *(const uint4*)g;
}
static __device__ inline void async_wait() {}
#endif

__constant__ float c_nf4[16] = {
    -1.0f, -0.6961928009986877f, -0.5250730514526367f, -0.39491748809814453f,
    -0.28444138169288635f, -0.18477343022823334f, -0.09105003625154495f, 0.0f,
    0.07958029955625534f, 0.16093020141124725f, 0.24611230194568634f,
    0.33791524171829224f, 0.44070982933044434f, 0.5626170039176941f,
    0.7229568362236023f, 1.0f};

static __device__ inline v8f wmma_f16(v16h a, v16h b, v8f c) {
    return __builtin_amdgcn_wmma_f32_16x16x32_f16(
        false, a, false, b, (short)0, c, false, false);
}

// A-fragment (16x32 f16): lane lo holds row M=lo; lanes 0-15 K=0..15,
// lanes 16-31 K=16..31  ->  two contiguous 16B chunks per lane.
static __device__ inline v16h load_afrag(const _Float16* rowbase, int lhi) {
    v16h a;
    ((uint4*)&a)[0] = *(const uint4*)(rowbase + lhi * 16);
    ((uint4*)&a)[1] = *(const uint4*)(rowbase + lhi * 16 + 8);
    return a;
}

// ---------------- f32 -> f16 convert ----------------
__global__ void k_cvt_f16(const float* __restrict__ x, _Float16* __restrict__ y, int n) {
    int i = blockIdx.x * blockDim.x + threadIdx.x;
    if (i < n) y[i] = (_Float16)x[i];
}

// ---------------- NF4 dequant + LoRA fold: Wt[k][n] = NF4[codes[n,k]]*absmax + (la@lb)[k,n]
__global__ void k_dequant_fold(const int* __restrict__ codes,
                               const float* __restrict__ absmax,
                               const float* __restrict__ la,   // [H,R]
                               const float* __restrict__ lb,   // [R,H]
                               _Float16* __restrict__ wt) {    // [H(k), H(n)]
    int idx = blockIdx.x * blockDim.x + threadIdx.x;   // 0 .. H*H-1
    int n = idx % H;
    int k = idx / H;
    int ci = n * H + k;                                 // codes row-major [out,in]
    float acc = c_nf4[codes[ci] & 15] * absmax[ci >> 6];
#pragma unroll
    for (int r = 0; r < RLORA; ++r)
        acc += la[k * RLORA + r] * lb[r * H + n];
    wt[(size_t)k * H + n] = (_Float16)acc;
}

// ---------------- f16 WMMA GEMM: out[M,N] = A[M,K] @ W[K,N] + bias ----------------
__global__ __launch_bounds__(256)
void k_gemm_wmma(const _Float16* __restrict__ A,    // [MROWS, H]
                 const _Float16* __restrict__ W,    // [H, H]
                 const float* __restrict__ bias,    // [H]
                 _Float16* __restrict__ outH,
                 float* __restrict__ outF,
                 int storeF) {
    __shared__ alignas(32) _Float16 As[128][32];     // A tile, row-major
    __shared__ alignas(32) _Float16 Bt[128][32];     // W tile, transposed [n][k]

    const int tid  = threadIdx.x;
    const int lane = tid & 31;
    const int wv   = tid >> 5;        // 0..7
    const int lo   = lane & 15;
    const int lhi  = lane >> 4;
    const int wy   = wv >> 2;         // 0..1  (64-row band)
    const int wx   = wv & 3;          // 0..3  (32-col band)
    const int bm   = blockIdx.y * 128;
    const int bn   = blockIdx.x * 128;

    const v8f vzero = {0.f,0.f,0.f,0.f,0.f,0.f,0.f,0.f};
    v8f acc[4][2];
#pragma unroll
    for (int mi = 0; mi < 4; ++mi)
#pragma unroll
        for (int ni = 0; ni < 2; ++ni) acc[mi][ni] = vzero;

    for (int k0 = 0; k0 < H; k0 += 32) {
        __syncthreads();
        {   // stage A: 128 rows x 32 halfs (async global -> LDS)
            int row = tid >> 1, ch = (tid & 1) * 16;
            const _Float16* src = A + (size_t)(bm + row) * H + k0 + ch;
            async_ld_b128(src,     &As[row][ch]);
            async_ld_b128(src + 8, &As[row][ch + 8]);
        }
        {   // stage W transposed: 32 k-rows x 128 n-cols -> Bt[n][k] (scatter)
            int kr = tid >> 3, nch = (tid & 7) * 16;
            const _Float16* src = W + (size_t)(k0 + kr) * H + bn + nch;
            _Float16 tmp[16];
            *(uint4*)&tmp[0] = *(const uint4*)src;
            *(uint4*)&tmp[8] = *(const uint4*)(src + 8);
#pragma unroll
            for (int i = 0; i < 16; ++i) Bt[nch + i][kr] = tmp[i];
        }
        if (k0 + 32 < H)   // hint next A tile toward the cache (global_prefetch_b8)
            __builtin_prefetch(A + (size_t)(bm + (tid >> 1)) * H + k0 + 32, 0, 1);
        async_wait();
        __syncthreads();

        v16h af[4];
#pragma unroll
        for (int mi = 0; mi < 4; ++mi)
            af[mi] = load_afrag(&As[wy * 64 + mi * 16 + lo][0], lhi);
        v16h bf[2];
#pragma unroll
        for (int ni = 0; ni < 2; ++ni)
            bf[ni] = *(const v16h*)&Bt[wx * 32 + ni * 16 + lo][lhi * 16];
#pragma unroll
        for (int mi = 0; mi < 4; ++mi)
#pragma unroll
            for (int ni = 0; ni < 2; ++ni)
                acc[mi][ni] = wmma_f16(af[mi], bf[ni], acc[mi][ni]);
    }

#pragma unroll
    for (int mi = 0; mi < 4; ++mi)
#pragma unroll
        for (int ni = 0; ni < 2; ++ni)
#pragma unroll
            for (int r = 0; r < 8; ++r) {
                int row = bm + wy * 64 + mi * 16 + lhi * 8 + r;
                int col = bn + wx * 32 + ni * 16 + lo;
                float v = acc[mi][ni][r] + bias[col];
                if (storeF) outF[(size_t)row * H + col] = v;
                else        outH[(size_t)row * H + col] = (_Float16)v;
            }
}

// ---------------- Flash attention with WMMA ----------------
__global__ __launch_bounds__(256)
void k_attn_wmma(const _Float16* __restrict__ Q,
                 const _Float16* __restrict__ Kd,
                 const _Float16* __restrict__ V,
                 const float* __restrict__ mask,   // [B,1,S,S]
                 _Float16* __restrict__ ctx) {     // [B,S,H]
    __shared__ alignas(32) _Float16 Ks[32][128];     // keys: [krow][dim]
    __shared__ alignas(32) _Float16 Vt[128][32];     // values transposed: [dim][krow]
    __shared__ alignas(32) _Float16 Ps[8][16][32];   // per-wave P staging

    const int tid  = threadIdx.x;
    const int lane = tid & 31;
    const int wv   = tid >> 5;
    const int lo   = lane & 15;
    const int lhi  = lane >> 4;
    const int qt   = blockIdx.x;          // 0..15
    const int hh   = blockIdx.y;          // 0..15
    const int bz   = blockIdx.z;          // 0..1
    const int hoff = hh * HD;
    const int qbase = qt * 128 + wv * 16;
    const float scale = 0.08838834764831845f;   // 1/sqrt(128)

    // Q band (16x128) as 4 A-fragments, loaded straight from global
    v16h qf[4];
#pragma unroll
    for (int kk = 0; kk < 4; ++kk)
        qf[kk] = load_afrag(Q + (size_t)(bz * SQ + qbase + lo) * H + hoff + kk * 32, lhi);

    const v8f vzero = {0.f,0.f,0.f,0.f,0.f,0.f,0.f,0.f};
    v8f o[8];
#pragma unroll
    for (int dt = 0; dt < 8; ++dt) o[dt] = vzero;
    float m_[8], l_[8];
#pragma unroll
    for (int r = 0; r < 8; ++r) { m_[r] = -1e30f; l_[r] = 0.f; }

    for (int kc = 0; kc < SQ; kc += 32) {
        __syncthreads();
        {   // stage 32 key rows (async global -> LDS)
            int kr = tid >> 3, ch = (tid & 7) * 16;
            const _Float16* src = Kd + (size_t)(bz * SQ + kc + kr) * H + hoff + ch;
            async_ld_b128(src,     &Ks[kr][ch]);
            async_ld_b128(src + 8, &Ks[kr][ch + 8]);
        }
        {   // stage 32 value rows, transposed (scatter)
            int kr = tid >> 3, ch = (tid & 7) * 16;
            const _Float16* src = V + (size_t)(bz * SQ + kc + kr) * H + hoff + ch;
            _Float16 tmp[16];
            *(uint4*)&tmp[0] = *(const uint4*)src;
            *(uint4*)&tmp[8] = *(const uint4*)(src + 8);
#pragma unroll
            for (int i = 0; i < 16; ++i) Vt[ch + i][kr] = tmp[i];
        }
        async_wait();
        __syncthreads();

        // scores: S = Q @ K^T  -> two 16x16 tiles (32 keys)
        v8f s0 = vzero, s1 = vzero;
#pragma unroll
        for (int kk = 0; kk < 4; ++kk) {
            v16h b0 = *(const v16h*)&Ks[lo][kk * 32 + lhi * 16];
            v16h b1 = *(const v16h*)&Ks[16 + lo][kk * 32 + lhi * 16];
            s0 = wmma_f16(qf[kk], b0, s0);
            s1 = wmma_f16(qf[kk], b1, s1);
        }

        // online softmax (per-row reductions across the 16 lanes of each half)
        float alpha[8];
#pragma unroll
        for (int r = 0; r < 8; ++r) {
            int qr = qbase + lhi * 8 + r;
            const float* mrow = mask + ((size_t)bz * SQ + qr) * SQ + kc;
            float v0 = s0[r] * scale + mrow[lo];
            float v1 = s1[r] * scale + mrow[16 + lo];
            float mx = fmaxf(v0, v1);
#pragma unroll
            for (int d = 8; d >= 1; d >>= 1) mx = fmaxf(mx, __shfl_xor(mx, d, 32));
            float mn = fmaxf(m_[r], mx);
            float a  = __expf(m_[r] - mn);
            float p0 = __expf(v0 - mn);
            float p1 = __expf(v1 - mn);
            float rs = p0 + p1;
#pragma unroll
            for (int d = 8; d >= 1; d >>= 1) rs += __shfl_xor(rs, d, 32);
            l_[r] = l_[r] * a + rs;
            m_[r] = mn;
            alpha[r] = a;
            Ps[wv][lhi * 8 + r][lo]      = (_Float16)p0;
            Ps[wv][lhi * 8 + r][16 + lo] = (_Float16)p1;
        }
#pragma unroll
        for (int dt = 0; dt < 8; ++dt)
#pragma unroll
            for (int r = 0; r < 8; ++r) o[dt][r] *= alpha[r];

        // O += P @ V  (P via LDS round-trip to A-fragment layout)
        v16h pf = load_afrag(&Ps[wv][lo][0], lhi);
#pragma unroll
        for (int dt = 0; dt < 8; ++dt) {
            v16h vb = *(const v16h*)&Vt[dt * 16 + lo][lhi * 16];
            o[dt] = wmma_f16(pf, vb, o[dt]);
        }
    }

#pragma unroll
    for (int dt = 0; dt < 8; ++dt)
#pragma unroll
        for (int r = 0; r < 8; ++r) {
            int row = bz * SQ + qbase + lhi * 8 + r;
            int col = hoff + dt * 16 + lo;
            ctx[(size_t)row * H + col] = (_Float16)(o[dt][r] / l_[r]);
        }
}

// ---------------- host launcher ----------------
extern "C" void kernel_launch(void* const* d_in, const int* in_sizes, int n_in,
                              void* d_out, int out_size, void* d_ws, size_t ws_size,
                              hipStream_t stream) {
    const float* x        = (const float*)d_in[0];
    const int*   codes[4] = {(const int*)d_in[1],  (const int*)d_in[6],
                             (const int*)d_in[11], (const int*)d_in[16]};
    const float* amax[4]  = {(const float*)d_in[2],  (const float*)d_in[7],
                             (const float*)d_in[12], (const float*)d_in[17]};
    const float* bias[4]  = {(const float*)d_in[3],  (const float*)d_in[8],
                             (const float*)d_in[13], (const float*)d_in[18]};
    const float* la[4]    = {(const float*)d_in[4],  (const float*)d_in[9],
                             (const float*)d_in[14], (const float*)d_in[19]};
    const float* lb[4]    = {(const float*)d_in[5],  (const float*)d_in[10],
                             (const float*)d_in[15], (const float*)d_in[20]};
    const float* mask     = (const float*)d_in[21];
    float* out            = (float*)d_out;

    // workspace layout (f16 buffers)
    char* ws = (char*)d_ws;
    const size_t XB = (size_t)MROWS * H * sizeof(_Float16);   // 16 MB
    const size_t WB = (size_t)H * H * sizeof(_Float16);       //  8 MB
    _Float16* xh = (_Float16*)(ws);
    _Float16* wh = (_Float16*)(ws + XB);
    _Float16* qh = (_Float16*)(ws + XB + WB);
    _Float16* kh = (_Float16*)(ws + XB + WB + XB);
    _Float16* vh = (_Float16*)(ws + XB + WB + 2 * XB);
    _Float16* ch = (_Float16*)(ws + XB + WB + 3 * XB);

    const int NX = MROWS * H;
    k_cvt_f16<<<(NX + 255) / 256, 256, 0, stream>>>(x, xh, NX);

    dim3 ggrid(H / 128, MROWS / 128);   // (16, 32)
    _Float16* proj[3] = {qh, kh, vh};
    for (int p = 0; p < 3; ++p) {
        k_dequant_fold<<<(H * H) / 256, 256, 0, stream>>>(codes[p], amax[p], la[p], lb[p], wh);
        k_gemm_wmma<<<ggrid, 256, 0, stream>>>(xh, wh, bias[p], proj[p], nullptr, 0);
    }

    k_attn_wmma<<<dim3(SQ / 128, HEADS, BB), 256, 0, stream>>>(qh, kh, vh, mask, ch);

    k_dequant_fold<<<(H * H) / 256, 256, 0, stream>>>(codes[3], amax[3], la[3], lb[3], wh);
    k_gemm_wmma<<<ggrid, 256, 0, stream>>>(ch, wh, bias[3], nullptr, out, 1);

    (void)in_sizes; (void)n_in; (void)out_size; (void)ws_size;
}
